// MultiHeadAttention_3238405341392
// MI455X (gfx1250) — compile-verified, matrix operand order checked by reading, not run on hardware
//
#include <hip/hip_runtime.h>

// ---------------------------------------------------------------------------
// MI455X / gfx1250 multi-head attention forward (wave32, bf16 WMMA)
// ---------------------------------------------------------------------------
constexpr int N_NODES = 8192;   // nodes
constexpr int EDIM    = 1024;   // hidden size
constexpr float QSCALE = 0.125f; // (E/HEADS)^-0.5 = 64^-0.5

typedef __attribute__((ext_vector_type(16))) __bf16 v16bf;
typedef __attribute__((ext_vector_type(8)))  float  v8f;

// K-offset pattern inside a 16-bit A/B fragment (16x32, wave32):
// VGPR v, lane-half h: k0 = 16*(v/4) + 8*h + 2*(v%4); elements k0, k0+1.
// Per lane this covers two contiguous 8-element chunks -> 2x b128 loads.
__device__ __forceinline__ int kpat(int v, int h) {
    return ((v >> 2) << 4) + (h << 3) + ((v & 3) << 1);
}

// A-style fragment from row-major bf16 (global or LDS). base -> (row m, col 0).
__device__ __forceinline__ v16bf load_abf(const __bf16* base, int h) {
    v16bf f;
#pragma unroll
    for (int v = 0; v < 8; ++v) {
        int k0 = kpat(v, h);
        union { unsigned int u; __bf16 e[2]; } t;
        t.u = *(const unsigned int*)(base + k0);   // 4B-aligned paired load
        f[2 * v]     = t.e[0];
        f[2 * v + 1] = t.e[1];
    }
    return f;
}

// A-style fragment from row-major f32, converting to bf16 in-register.
__device__ __forceinline__ v16bf load_af32(const float* base, int h) {
    v16bf f;
#pragma unroll
    for (int v = 0; v < 8; ++v) {
        int k0 = kpat(v, h);
        f[2 * v]     = (__bf16)base[k0];
        f[2 * v + 1] = (__bf16)base[k0 + 1];
    }
    return f;
}

__device__ __forceinline__ v8f wmma_bf16(v16bf a, v16bf b, v8f c) {
    return __builtin_amdgcn_wmma_f32_16x16x32_bf16(
        /*neg_a=*/false, a, /*neg_b=*/false, b,
        /*c_mod=*/(short)0, c, /*reuse_a=*/false, /*reuse_b=*/false);
}

__device__ __forceinline__ float rmax16(float x) {
#pragma unroll
    for (int i = 1; i < 16; i <<= 1) x = fmaxf(x, __shfl_xor(x, i, 32));
    return x;
}
__device__ __forceinline__ float rsum16(float x) {
#pragma unroll
    for (int i = 1; i < 16; i <<= 1) x += __shfl_xor(x, i, 32);
    return x;
}

// ---------------------------------------------------------------------------
// Weight transpose + f32->bf16:  WT[n*E+k] = W[k*E+n]
// ---------------------------------------------------------------------------
__global__ __launch_bounds__(256) void wtrans_kernel(const float* __restrict__ W,
                                                     __bf16* __restrict__ WT) {
    int idx = blockIdx.x * 256 + threadIdx.x;   // idx = k*E + n (coalesced read)
    int n = idx & (EDIM - 1);
    int k = idx >> 10;
    WT[(size_t)n * EDIM + k] = (__bf16)W[idx];
}

// ---------------------------------------------------------------------------
// C = (A @ B + bias) * scale      A: f32 [M,K] row-major
//                                 BT: bf16 [Nc,K] (i.e. B transposed)
// Block: 256 threads = 8 waves, computes 64x128 C tile; wave = 32x32.
// OUT_MODE: 0 = f32 row-major, 1 = bf16 row-major, 2 = bf16 TRANSPOSED
//           (Cb[col*M + row], used for V so attention streams vT rows).
// ---------------------------------------------------------------------------
template <int OUT_MODE>
__global__ __launch_bounds__(256) void gemm_bias_kernel(
    const float* __restrict__ A, const __bf16* __restrict__ BT,
    const float* __restrict__ bias, float scale,
    float* __restrict__ Cf, __bf16* __restrict__ Cb,
    int M, int Nc, int K) {
    const int tid  = threadIdx.x;
    const int w    = tid >> 5;
    const int lane = tid & 31;
    const int m    = lane & 15;
    const int h    = lane >> 4;
    const int row0 = blockIdx.y * 64 + (w >> 2) * 32;
    const int col0 = blockIdx.x * 128 + (w & 3) * 32;

    v8f acc[2][2];
#pragma unroll
    for (int t = 0; t < 2; ++t)
#pragma unroll
        for (int u = 0; u < 2; ++u) acc[t][u] = (v8f){};

    for (int kk = 0; kk < K; kk += 32) {
        v16bf a0 = load_af32(A + (size_t)(row0 + m) * K + kk, h);
        v16bf a1 = load_af32(A + (size_t)(row0 + 16 + m) * K + kk, h);
        v16bf b0 = load_abf(BT + (size_t)(col0 + m) * K + kk, h);
        v16bf b1 = load_abf(BT + (size_t)(col0 + 16 + m) * K + kk, h);
        acc[0][0] = wmma_bf16(a0, b0, acc[0][0]);
        acc[0][1] = wmma_bf16(a0, b1, acc[0][1]);
        acc[1][0] = wmma_bf16(a1, b0, acc[1][0]);
        acc[1][1] = wmma_bf16(a1, b1, acc[1][1]);
    }

#pragma unroll
    for (int u = 0; u < 2; ++u) {
        const int col = col0 + u * 16 + m;
        const float bv = bias[col];
#pragma unroll
        for (int t = 0; t < 2; ++t) {
#pragma unroll
            for (int r = 0; r < 8; ++r) {
                const int row = row0 + t * 16 + r + (h << 3);
                const float val = (acc[t][u][r] + bv) * scale;
                if (OUT_MODE == 0)      Cf[(size_t)row * Nc + col] = val;
                else if (OUT_MODE == 1) Cb[(size_t)row * Nc + col] = (__bf16)val;
                else                    Cb[(size_t)col * M  + row] = (__bf16)val;
            }
        }
    }
}

// ---------------------------------------------------------------------------
// Flash attention: O = softmax(qb @ kb^T + bias) @ v, with V given TRANSPOSED
// as vT[E][N] bf16 so V fragments are contiguous A-style (b128) loads.
// qb pre-scaled by QSCALE. Block = 8 waves = 32 query rows, full E strip.
// Per 128-key block: wave w owns score key-strip [16w,16w+16) and output
// column strip [128w, 128w+128).
// ---------------------------------------------------------------------------
__global__ __launch_bounds__(256) void attn_fwd(
    const __bf16* __restrict__ qb, const __bf16* __restrict__ kb,
    const __bf16* __restrict__ vT, const float* __restrict__ bias,
    float* __restrict__ O) {
    constexpr int MQ = 32, BK = 128;
    __shared__ __bf16 Ps[MQ][BK];          // post-softmax probs (bf16)
    __shared__ float  red[8][MQ];          // per-wave row partials (max / sum)
    __shared__ float  Mrun[MQ], Lrun[MQ], Alph[MQ], Mnew[MQ];

    const int tid  = threadIdx.x;
    const int w    = tid >> 5;
    const int lane = tid & 31;
    const int m    = lane & 15;           // C-layout: n index / A-layout: row
    const int h    = lane >> 4;
    const int q0   = blockIdx.x * MQ;
    const int ks   = w * 16;              // key strip within block
    const int c0   = w * 128;             // output column strip

    if (tid < MQ) { Mrun[tid] = -3.0e38f; Lrun[tid] = 0.0f; }

    v8f acc[2][8];
#pragma unroll
    for (int t = 0; t < 2; ++t)
#pragma unroll
        for (int nt = 0; nt < 8; ++nt) acc[t][nt] = (v8f){};
    __syncthreads();

    for (int kb0 = 0; kb0 < N_NODES; kb0 += BK) {
        // Prefetch next key-block K strip (gfx1250 global_prefetch_b8 path).
        if (kb0 + BK < N_NODES) {
            const __bf16* nk = kb + (size_t)(kb0 + BK + ks) * EDIM + lane * 64;
            __builtin_prefetch(nk, 0, 1);
        }
        // ---- scores: S = qb(32xE) @ kb_strip^T, full-E contraction ----
        v8f s0 = (v8f){}, s1 = (v8f){};
        for (int e0 = 0; e0 < EDIM; e0 += 32) {
            v16bf a0 = load_abf(qb + (size_t)(q0 + m) * EDIM + e0, h);
            v16bf a1 = load_abf(qb + (size_t)(q0 + 16 + m) * EDIM + e0, h);
            v16bf kf = load_abf(kb + (size_t)(kb0 + ks + m) * EDIM + e0, h);
            s0 = wmma_bf16(a0, kf, s0);
            s1 = wmma_bf16(a1, kf, s1);
        }
        // ---- add bias, per-row max over this wave's 16-key strip ----
#pragma unroll
        for (int r = 0; r < 8; ++r) {
            s0[r] += bias[(size_t)(q0 + r + (h << 3)) * N_NODES + kb0 + ks + m];
            s1[r] += bias[(size_t)(q0 + 16 + r + (h << 3)) * N_NODES + kb0 + ks + m];
        }
#pragma unroll
        for (int r = 0; r < 8; ++r) {
            float mx0 = rmax16(s0[r]);
            float mx1 = rmax16(s1[r]);
            if (m == 0) {
                red[w][r + (h << 3)]      = mx0;
                red[w][16 + r + (h << 3)] = mx1;
            }
        }
        __syncthreads();
        if (tid < MQ) {
            float bm = red[0][tid];
#pragma unroll
            for (int ww = 1; ww < 8; ++ww) bm = fmaxf(bm, red[ww][tid]);
            const float mo = Mrun[tid];
            const float mn = fmaxf(mo, bm);
            Mnew[tid] = mn;
            Alph[tid] = __expf(mo - mn);
            Mrun[tid] = mn;
        }
        __syncthreads();
        // ---- P = exp(S - Mnew), row partial sums, write P to LDS ----
#pragma unroll
        for (int r = 0; r < 8; ++r) {
            float p0 = __expf(s0[r] - Mnew[r + (h << 3)]);
            float p1 = __expf(s1[r] - Mnew[16 + r + (h << 3)]);
            Ps[r + (h << 3)][ks + m]      = (__bf16)p0;
            Ps[16 + r + (h << 3)][ks + m] = (__bf16)p1;
            float sm0 = rsum16(p0);
            float sm1 = rsum16(p1);
            if (m == 0) {
                red[w][r + (h << 3)]      = sm0;
                red[w][16 + r + (h << 3)] = sm1;
            }
        }
        __syncthreads();
        if (tid < MQ) {
            float s = 0.0f;
#pragma unroll
            for (int ww = 0; ww < 8; ++ww) s += red[ww][tid];
            Lrun[tid] = Lrun[tid] * Alph[tid] + s;
        }
        // ---- rescale accumulator by alpha ----
#pragma unroll
        for (int r = 0; r < 8; ++r) {
            const float aa = Alph[r + (h << 3)];
            const float ab = Alph[16 + r + (h << 3)];
#pragma unroll
            for (int nt = 0; nt < 8; ++nt) {
                acc[0][nt][r] *= aa;
                acc[1][nt][r] *= ab;
            }
        }
        // ---- O += P(32x128) @ V(128 x col-strip), V from vT rows ----
        for (int kk = 0; kk < BK; kk += 32) {
            v16bf p0 = load_abf(&Ps[m][kk], h);
            v16bf p1 = load_abf(&Ps[16 + m][kk], h);
#pragma unroll
            for (int nt = 0; nt < 8; ++nt) {
                // B frag of V: element (k=key, n=e) = vT[e*N + key]
                v16bf vf = load_abf(
                    vT + (size_t)(c0 + nt * 16 + m) * N_NODES + kb0 + kk, h);
                acc[0][nt] = wmma_bf16(p0, vf, acc[0][nt]);
                acc[1][nt] = wmma_bf16(p1, vf, acc[1][nt]);
            }
        }
        __syncthreads();   // protects Ps/red/Lrun for next iteration
    }

    // ---- finalize: divide by running sum, store f32 ----
#pragma unroll
    for (int r = 0; r < 8; ++r) {
        const float li0 = 1.0f / Lrun[r + (h << 3)];
        const float li1 = 1.0f / Lrun[16 + r + (h << 3)];
#pragma unroll
        for (int nt = 0; nt < 8; ++nt) {
            O[(size_t)(q0 + r + (h << 3)) * EDIM + c0 + nt * 16 + m]      = acc[0][nt][r] * li0;
            O[(size_t)(q0 + 16 + r + (h << 3)) * EDIM + c0 + nt * 16 + m] = acc[1][nt][r] * li1;
        }
    }
}

// ---------------------------------------------------------------------------
// Host launcher
// ---------------------------------------------------------------------------
extern "C" void kernel_launch(void* const* d_in, const int* in_sizes, int n_in,
                              void* d_out, int out_size, void* d_ws, size_t ws_size,
                              hipStream_t stream) {
    const float* q    = (const float*)d_in[0];
    const float* k    = (const float*)d_in[1];
    const float* v    = (const float*)d_in[2];
    const float* bias = (const float*)d_in[3];
    const float* Wq   = (const float*)d_in[4];
    const float* bq   = (const float*)d_in[5];
    const float* Wk   = (const float*)d_in[6];
    const float* bk   = (const float*)d_in[7];
    const float* Wv   = (const float*)d_in[8];
    const float* bv   = (const float*)d_in[9];
    const float* Wo   = (const float*)d_in[10];
    const float* bo   = (const float*)d_in[11];

    char* p = (char*)d_ws;
    auto take = [&](size_t bytes) -> void* {
        void* r = (void*)p;
        p += (bytes + 255) & ~(size_t)255;
        return r;
    };
    const size_t WB = (size_t)EDIM * EDIM * sizeof(__bf16);
    const size_t NB = (size_t)N_NODES * EDIM * sizeof(__bf16);
    __bf16* WqT = (__bf16*)take(WB);
    __bf16* WkT = (__bf16*)take(WB);
    __bf16* WvT = (__bf16*)take(WB);
    __bf16* WoT = (__bf16*)take(WB);
    __bf16* qb  = (__bf16*)take(NB);
    __bf16* kbm = (__bf16*)take(NB);
    __bf16* vTb = (__bf16*)take(NB);   // V projected, stored transposed [E][N]
    float*  Oat = (float*)take((size_t)N_NODES * EDIM * sizeof(float));

    // 1) weights -> bf16 transposed
    const int wgrid = EDIM * EDIM / 256;
    wtrans_kernel<<<wgrid, 256, 0, stream>>>(Wq, WqT);
    wtrans_kernel<<<wgrid, 256, 0, stream>>>(Wk, WkT);
    wtrans_kernel<<<wgrid, 256, 0, stream>>>(Wv, WvT);
    wtrans_kernel<<<wgrid, 256, 0, stream>>>(Wo, WoT);

    // 2) projections (bf16 outputs; q pre-scaled by 1/8; V transposed)
    dim3 ggrid(EDIM / 128, N_NODES / 64);
    gemm_bias_kernel<1><<<ggrid, 256, 0, stream>>>(q, WqT, bq, QSCALE,
                                                   nullptr, qb, N_NODES, EDIM, EDIM);
    gemm_bias_kernel<1><<<ggrid, 256, 0, stream>>>(k, WkT, bk, 1.0f,
                                                   nullptr, kbm, N_NODES, EDIM, EDIM);
    gemm_bias_kernel<2><<<ggrid, 256, 0, stream>>>(v, WvT, bv, 1.0f,
                                                   nullptr, vTb, N_NODES, EDIM, EDIM);

    // 3) flash attention
    attn_fwd<<<N_NODES / 32, 256, 0, stream>>>(qb, kbm, vTb, bias, Oat);

    // 4) output projection -> f32 d_out
    gemm_bias_kernel<0><<<ggrid, 256, 0, stream>>>(Oat, WoT, bo, 1.0f,
                                                   (float*)d_out, nullptr,
                                                   N_NODES, EDIM, EDIM);
}
